// FuturePredictionODE_59176059404993
// MI455X (gfx1250) — compile-verified, matrix operand order checked by reading, not run on hardware
//
#include <hip/hip_runtime.h>
#include <hip/hip_bf16.h>

// ---------------------------------------------------------------------------
// FuturePredictionODE for MI455X (gfx1250, wave32, WMMA + TDM).
// All convolutions run as implicit GEMM through v_wmma_f32_16x16x32_bf16
// (bf16 inputs, f32 accumulate). Weights converted once per call to a
// [tap][co][ci] bf16 layout; the 64x32 B-tile is DMA'd into LDS by the
// Tensor Data Mover (tensor_load_to_lds + s_wait_tensorcnt) with TDM LDS
// padding producing the bank-friendly 48-element row stride. Conv kernel is
// templated on <CIN,TAPS> (unrolled K chunks, many static v_wmma); each wave
// register-blocks a 32x16 output (2 accumulators sharing one B fragment).
// ---------------------------------------------------------------------------

typedef __attribute__((ext_vector_type(16))) __bf16 v16bf;
typedef __attribute__((ext_vector_type(8)))  __bf16 v8bf;
typedef __attribute__((ext_vector_type(8)))  float  v8f;
typedef __attribute__((ext_vector_type(4)))  unsigned int v4u;
typedef __attribute__((ext_vector_type(8)))  int v8i;
typedef __attribute__((ext_vector_type(4)))  int v4i;

#define DEVFN __device__ __forceinline__

constexpr int B_   = 2;
constexpr int NC_  = 3;
constexpr int NL_  = 3;
constexpr int C_   = 64;
constexpr int F_   = 6;
constexpr int HH   = 48;
constexpr int WW   = 48;
constexpr int HID_ = 128;
constexpr int KST  = 31;          // K time steps
constexpr float DT_ = 0.1f;
constexpr int IMG  = HH * WW;     // 2304 spatial positions
constexpr int CI   = C_ * IMG;    // one 64-ch image plane (floats)

enum { ACT_NONE = 0, ACT_SIG = 1, ACT_TANH = 2, ACT_RELU = 3 };

// ---------------------------------------------------------------------------
// Schedule kernel: replicates argsort / obs-step / pred-step logic on device
// (timestamps are device inputs; must stay on-stream for graph capture).
// ---------------------------------------------------------------------------
__global__ void k_schedule(const float* __restrict__ cam_ts,
                           const float* __restrict__ lid_ts,
                           const float* __restrict__ tgt,
                           int* __restrict__ has_obs,
                           int* __restrict__ srcsel,
                           int* __restrict__ psel) {
  if (threadIdx.x != 0 || blockIdx.x != 0) return;
  for (int b = 0; b < B_; ++b) {
    float ts[6];
    for (int i = 0; i < NC_; ++i) ts[i]       = cam_ts[b * NC_ + i];
    for (int i = 0; i < NL_; ++i) ts[NC_ + i] = lid_ts[b * NL_ + i];
    float t0 = ts[0];
    for (int i = 1; i < 6; ++i) t0 = fminf(t0, ts[i]);
    float rel[6];
    for (int i = 0; i < 6; ++i) rel[i] = ts[i] - t0;
    // stable insertion sort (ascending) -> order
    int ord[6];
    for (int i = 0; i < 6; ++i) ord[i] = i;
    for (int i = 1; i < 6; ++i) {
      int oi = ord[i]; float v = rel[oi]; int j = i - 1;
      while (j >= 0 && rel[ord[j]] > v) { ord[j + 1] = ord[j]; --j; }
      ord[j + 1] = oi;
    }
    int ho[KST], oi[KST];
    for (int k = 0; k < KST; ++k) { ho[k] = 0; oi[k] = 0; }
    for (int i = 0; i < 6; ++i) {
      float t = rel[ord[i]];
      int st = (int)ceilf(t / DT_) - 1;
      st = st < 0 ? 0 : (st > KST - 1 ? KST - 1 : st);
      ho[st] = 1; oi[st] = i;                 // last writer wins (matches .set)
    }
    for (int k = 0; k < KST; ++k) {
      has_obs[b * KST + k] = ho[k];
      srcsel[b * KST + k]  = ord[oi[k]];      // slot into concat(cam,lid)
    }
    float trel = tgt[b] - t0;
    int ps[KST];
    for (int k = 0; k < KST; ++k) ps[k] = -1;
    for (int j = 0; j < F_; ++j) {
      float Tj = (float)(j + 1) * (trel / (float)F_) + 0.0001f;
      int st = (int)ceilf(Tj / DT_) - 1;
      st = st < 0 ? 0 : (st > KST - 1 ? KST - 1 : st);
      ps[st] = j;
    }
    for (int k = 0; k < KST; ++k) psel[b * KST + k] = ps[k];
  }
}

// ---------------------------------------------------------------------------
// Weight prep: fp32 OIHW -> bf16 [tap][co][ci] (optionally a ci sub-range).
// Rows = output channels with contiguous K: TDM-tileable, and manual staging
// reads 16B-contiguous runs.
// ---------------------------------------------------------------------------
__global__ void k_prep_w(const float* __restrict__ src, __bf16* __restrict__ dst,
                         int Co, int Ci, int ciOff, int CinTot, int T) {
  int i = blockIdx.x * blockDim.x + threadIdx.x;
  int total = T * Ci * Co;
  if (i >= total) return;
  int ci = i % Ci;
  int r  = i / Ci;
  int co = r % Co;
  int t  = r / Co;
  dst[i] = (__bf16)src[((long)co * CinTot + ciOff + ci) * T + t];
}

// ---------------------------------------------------------------------------
// Generic implicit-GEMM conv via WMMA, templated on <CIN, TAPS>.
// Block: 256 threads = 8 waves. Block tile: 64 spatial (M) x 64 channels (N).
// Waves arranged 2(M) x 4(N); each wave computes a 32x16 tile with two f32
// accumulators sharing one B fragment -> 2 v_wmma per K-chunk, and the
// channel-chunk loop is unrolled (compile-time trip count).
// B tile (64 rows x 32 k, bf16) staged by the Tensor Data Mover when the
// toolchain exposes tensor_load_to_lds; else by manual loads.
// Grid: (IMG/64, Cout/64, nImages).
// ---------------------------------------------------------------------------
template <int CIN, int TAPS>
__global__ __launch_bounds__(256)
void k_conv_wmma(const float* __restrict__ X, long istride,
                 float* __restrict__ Y, long ostride,
                 const __bf16* __restrict__ Wp, const float* __restrict__ bias,
                 int Cout, int dil, int act, int biasPerImg) {
  __shared__ __bf16 ldsA[64 * 48];        // [row][k], row stride 48 (16B aligned)
  __shared__ __bf16 ldsB[64 * 48 + 16];   // [n][k], stride 48 (TDM pad) + slack

  const int tid   = threadIdx.x;
  const int img   = blockIdx.z;
  const int mBase = blockIdx.x * 64;
  const int nBase = blockIdx.y * 64;
  const float* Xi = X + (long)img * istride;
  float*       Yi = Y + (long)img * ostride;

  const int wave = tid >> 5, lane = tid & 31;
  const int wm = wave & 1;            // M group of 32 rows
  const int wn = wave >> 1;           // N group of 16 channels (0..3)
  const int half = lane >> 4, l16 = lane & 15;

  // staging assignments
  const int a_kk = tid >> 3;          // 0..31 : channel within chunk
  const int a_r0 = (tid & 7) * 8;     // 8 consecutive spatial rows
  const int b_n  = tid >> 2;          // output channel within tile (0..63)
  const int b_k0 = (tid & 3) * 8;     // 8 contiguous k elements

  v8f acc0, acc1;
  #pragma unroll
  for (int i = 0; i < 8; ++i) { acc0[i] = 0.0f; acc1[i] = 0.0f; }

  constexpr int CHUNKS = CIN / 32;
  constexpr int UC = (CHUNKS >= 4) ? 4 : CHUNKS;   // unroll factor

  for (int t = 0; t < TAPS; ++t) {
    int dy = 0, dx = 0;
    if (TAPS == 9) { dy = (t / 3 - 1) * dil; dx = (t % 3 - 1) * dil; }
    for (int cb = 0; cb < CHUNKS; cb += UC) {
      #pragma unroll
      for (int u = 0; u < UC; ++u) {
        const int ci0 = (cb + u) * 32;
        // ---- stage A tile (im2col patch, fp32 -> bf16) ----
        const float* Xc = Xi + (long)(ci0 + a_kk) * IMG;
        if (ci0 + 32 < CIN) __builtin_prefetch(Xc + 32 * IMG, 0, 0);
        #pragma unroll
        for (int j = 0; j < 8; ++j) {
          int m = mBase + a_r0 + j;
          int y = m / WW + dy;
          int x = m % WW + dx;
          float v = 0.0f;
          if ((unsigned)y < (unsigned)HH && (unsigned)x < (unsigned)WW)
            v = Xc[y * WW + x];
          ldsA[(a_r0 + j) * 48 + a_kk] = (__bf16)v;
        }
        // ---- stage B tile: weights [tap][co][ci], tile 64 rows x 32 k ----
#if __has_builtin(__builtin_amdgcn_tensor_load_to_lds)
        if (wave == 0) {
          // TDM descriptor (ISA 8.3/8.4): 2D tensor [Cout*TAPS][CIN],
          // tile [64][32] bf16, LDS pad 8 DWORDs after every 16 DWORDs
          // -> LDS row stride 48 elements.
          const unsigned rows = (unsigned)(Cout * TAPS);
          unsigned long long ga = (unsigned long long)(const void*)
              (Wp + ((long)t * Cout + nBase) * CIN + ci0);
          unsigned ldsOff = (unsigned)(unsigned long long)(&ldsB[0]);
          v4u g0;
          g0[0] = 1u;                                   // count=1 (valid D#)
          g0[1] = ldsOff;                               // lds_addr
          g0[2] = (unsigned)ga;                         // global_addr[31:0]
          g0[3] = (unsigned)((ga >> 32) & 0x1FFFFFFu)   // global_addr[56:32]
                  | (2u << 30);                         // type=2 ("image")
          v8i g1;
          g1[0] = (int)((1u << 16)      // data_size = 2 bytes
                      | (1u << 20)      // pad_enable
                      | (3u << 22)      // pad_interval: 16 DWORDs (64B)
                      | (7u << 25));    // pad_amount: 8 DWORDs (32B)
          g1[1] = (int)(((unsigned)CIN & 0xFFFFu) << 16);        // dim0[15:0]
          g1[2] = (int)((((unsigned)CIN >> 16) & 0xFFFFu)
                      | ((rows & 0xFFFFu) << 16));               // dim0[31:16], dim1[15:0]
          g1[3] = (int)(((rows >> 16) & 0xFFFFu) | (32u << 16)); // dim1[31:16], tile_dim0=32
          g1[4] = 64;                                            // tile_dim1=64, tile_dim2=0
          g1[5] = (int)CIN;                                      // dim0_stride[31:0]
          g1[6] = 0;
          g1[7] = 0;
          v4i z4; z4[0] = 0; z4[1] = 0; z4[2] = 0; z4[3] = 0;
#if defined(__clang_major__) && (__clang_major__ >= 23)
          v8i z8;
          #pragma unroll
          for (int q = 0; q < 8; ++q) z8[q] = 0;
          __builtin_amdgcn_tensor_load_to_lds(g0, g1, z4, z4, z8, 0);
#else
          __builtin_amdgcn_tensor_load_to_lds(g0, g1, z4, z4, 0);
#endif
          __builtin_amdgcn_s_wait_tensorcnt(0);
        }
#else
        {
          const __bf16* Wc = Wp + ((long)(t * Cout + nBase + b_n)) * CIN + ci0 + b_k0;
          #pragma unroll
          for (int j = 0; j < 8; ++j)
            ldsB[b_n * 48 + b_k0 + j] = Wc[j];
        }
#endif
        __syncthreads();

        // ---- fragments (ISA 7.12.2 16-bit layouts) ----
        const __bf16* pa0 = &ldsA[(wm * 32 + l16) * 48 + half * 8];
        const __bf16* pa1 = pa0 + 16 * 48;
        const __bf16* pb  = &ldsB[(wn * 16 + l16) * 48 + half * 8];
        v8bf a00 = *(const v8bf*)pa0;
        v8bf a01 = *(const v8bf*)(pa0 + 16);
        v8bf a10 = *(const v8bf*)pa1;
        v8bf a11 = *(const v8bf*)(pa1 + 16);
        v8bf b0  = *(const v8bf*)pb;
        v8bf b1  = *(const v8bf*)(pb + 16);
        v16bf va0, va1, vb;
        #pragma unroll
        for (int e = 0; e < 8; ++e) {
          va0[e] = a00[e]; va0[e + 8] = a01[e];
          va1[e] = a10[e]; va1[e + 8] = a11[e];
          vb[e]  = b0[e];  vb[e + 8]  = b1[e];
        }
        acc0 = __builtin_amdgcn_wmma_f32_16x16x32_bf16(
            false, va0, false, vb, (short)0, acc0, false, false);
        acc1 = __builtin_amdgcn_wmma_f32_16x16x32_bf16(
            false, va1, false, vb, (short)0, acc1, false, false);
        __syncthreads();
      }
    }
  }

  // ---- epilogue: bias + activation, D layout per ISA (lane = N column) ----
  const int n = nBase + wn * 16 + l16;
  float bv = 0.0f;
  if (bias) bv = bias[(biasPerImg ? img * Cout : 0) + n];
  const int mrow0 = mBase + wm * 32 + half * 8;
  #pragma unroll
  for (int r = 0; r < 8; ++r) {
    float v0 = acc0[r] + bv;
    float v1 = acc1[r] + bv;
    if      (act == ACT_SIG)  { v0 = 1.0f / (1.0f + __expf(-v0));
                                v1 = 1.0f / (1.0f + __expf(-v1)); }
    else if (act == ACT_TANH) { v0 = tanhf(v0); v1 = tanhf(v1); }
    else if (act == ACT_RELU) { v0 = fmaxf(v0, 0.0f); v1 = fmaxf(v1, 0.0f); }
    Yi[(long)n * IMG + mrow0 + r]      = v0;
    Yi[(long)n * IMG + mrow0 + 16 + r] = v1;
  }
}

// ---------------------------------------------------------------------------
// Elementwise / glue kernels
// ---------------------------------------------------------------------------
DEVFN const float* obs_ptr(const float* cam, const float* lid, int b, int slot) {
  return (slot < NC_) ? cam + (long)(b * NC_ + slot) * CI
                      : lid + (long)(b * NL_ + (slot - NC_)) * CI;
}

__global__ void k_mul(float* __restrict__ o, const float* __restrict__ a,
                      const float* __restrict__ b, int n) {
  int i = blockIdx.x * blockDim.x + threadIdx.x;
  if (i < n) o[i] = a[i] * b[i];
}

__global__ void k_gruode(float* __restrict__ h, const float* __restrict__ z,
                         const float* __restrict__ g, int n) {
  int i = blockIdx.x * blockDim.x + threadIdx.x;
  if (i < n) h[i] += DT_ * (1.0f - z[i]) * (g[i] - h[i]);
}

__global__ void k_joint(float* __restrict__ h, const float* __restrict__ zj,
                        const float* __restrict__ gj,
                        const int* __restrict__ has_obs, int k) {
  int i = blockIdx.x * blockDim.x + threadIdx.x;
  if (i >= B_ * CI) return;
  int b = i / CI;
  if (has_obs[b * KST + k])
    h[i] = (1.0f - zj[i]) * h[i] + zj[i] * gj[i];
}

__global__ void k_build_xh(float* __restrict__ xh, const float* __restrict__ cam,
                           const float* __restrict__ lid, const float* __restrict__ h,
                           const int* __restrict__ srcsel, int k) {
  int i = blockIdx.x * blockDim.x + threadIdx.x;
  if (i >= B_ * 2 * CI) return;
  int b = i / (2 * CI), r = i % (2 * CI);
  float v;
  if (r < CI) v = obs_ptr(cam, lid, b, srcsel[b * KST + k])[r];
  else        v = h[b * CI + (r - CI)];
  xh[i] = v;
}

__global__ void k_xh_second(float* __restrict__ xh, const float* __restrict__ rj,
                            const float* __restrict__ h) {
  int i = blockIdx.x * blockDim.x + threadIdx.x;
  if (i >= B_ * CI) return;
  int b = i / CI, r = i % CI;
  xh[(long)b * 2 * CI + CI + r] = rj[i] * h[i];
}

__global__ void k_loss(const float* __restrict__ p, const float* __restrict__ cam,
                       const float* __restrict__ lid,
                       const int* __restrict__ has_obs,
                       const int* __restrict__ srcsel, int k,
                       float* __restrict__ loss) {
  int b = blockIdx.y;
  if (!has_obs[b * KST + k]) return;
  int i = blockIdx.x * 256 + threadIdx.x;
  float d = 0.0f;
  if (i < CI) {
    const float* xo = obs_ptr(cam, lid, b, srcsel[b * KST + k]);
    float t = p[(long)b * CI + i] - xo[i];
    d = t * t;
  }
  __shared__ float red[256];
  red[threadIdx.x] = d;
  __syncthreads();
  for (int s = 128; s > 0; s >>= 1) {
    if (threadIdx.x < s) red[threadIdx.x] += red[threadIdx.x + s];
    __syncthreads();
  }
  if (threadIdx.x == 0) atomicAdd(loss, red[0] * (1.0f / (float)CI));
}

__global__ void k_scatter_pred(float* __restrict__ preds, const float* __restrict__ p,
                               const int* __restrict__ psel, int k) {
  int i = blockIdx.x * blockDim.x + threadIdx.x;
  if (i >= B_ * CI) return;
  int b = i / CI, r = i % CI;
  int j = psel[b * KST + k];
  if (j >= 0) preds[(long)(b * F_ + j) * CI + r] = p[(long)b * CI + r];
}

__global__ void k_copy_t0(float* __restrict__ s, const float* __restrict__ seq) {
  int i = blockIdx.x * blockDim.x + threadIdx.x;
  if (i >= B_ * CI) return;
  int b = i / CI, r = i % CI;
  s[i] = seq[(long)b * F_ * CI + r];      // seq[b][0]
}

// xs = concat([seq[:,t], mode ? (1-r)*s : s], ch)
__global__ void k_build_xs(float* __restrict__ xs, const float* __restrict__ seq,
                           int t, const float* __restrict__ s,
                           const float* __restrict__ r, int mode) {
  int i = blockIdx.x * blockDim.x + threadIdx.x;
  if (i >= B_ * 2 * CI) return;
  int b = i / (2 * CI), q = i % (2 * CI);
  float v;
  if (q < CI) v = seq[(long)(b * F_ + t) * CI + q];
  else {
    int q2 = q - CI;
    float sv = s[(long)b * CI + q2];
    v = mode ? (1.0f - r[(long)b * CI + q2]) * sv : sv;
  }
  xs[i] = v;
}

__global__ void k_sgru_update(float* __restrict__ s, float* __restrict__ out_seq,
                              int t, const float* __restrict__ u,
                              const float* __restrict__ st) {
  int i = blockIdx.x * blockDim.x + threadIdx.x;
  if (i >= B_ * CI) return;
  int b = i / CI, r = i % CI;
  float v = (1.0f - u[i]) * s[i] + u[i] * st[i];
  s[i] = v;
  out_seq[(long)(b * F_ + t) * CI + r] = v;
}

__global__ void k_mean(const float* __restrict__ y, float* __restrict__ mean) {
  int img = blockIdx.x, c = threadIdx.x;                 // <<<12, 64>>>
  const float* p = y + ((long)img * C_ + c) * IMG;
  float s = 0.0f;
  for (int i = 0; i < IMG; ++i) s += p[i];
  mean[img * C_ + c] = s * (1.0f / (float)IMG);
}

__global__ void k_gp(const float* __restrict__ mean, const float* __restrict__ wgp,
                     const float* __restrict__ bgp, float* __restrict__ gp) {
  int i = blockIdx.x * blockDim.x + threadIdx.x;
  if (i >= B_ * F_ * HID_) return;
  int img = i / HID_, o = i % HID_;
  float s = bgp[o];
  for (int ci = 0; ci < C_; ++ci) s += wgp[o * C_ + ci] * mean[img * C_ + ci];
  gp[i] = fmaxf(s, 0.0f);
}

// zbias[img][o] = bproj[o] + sum_ci wproj[o][512+ci] * gp[img][ci]
__global__ void k_zbias(const float* __restrict__ gp, const float* __restrict__ wproj,
                        const float* __restrict__ bproj, float* __restrict__ zb) {
  int i = blockIdx.x * blockDim.x + threadIdx.x;
  if (i >= B_ * F_ * HID_) return;
  int img = i / HID_, o = i % HID_;
  float s = bproj[o];
  for (int ci = 0; ci < HID_; ++ci)
    s += wproj[(long)o * (5 * HID_) + 4 * HID_ + ci] * gp[img * HID_ + ci];
  zb[i] = s;
}

// ---------------------------------------------------------------------------
// Host orchestration
// ---------------------------------------------------------------------------
extern "C" void kernel_launch(void* const* d_in, const int* in_sizes, int n_in,
                              void* d_out, int out_size, void* d_ws, size_t ws_size,
                              hipStream_t stream) {
  (void)in_sizes; (void)n_in; (void)out_size; (void)ws_size;

  // Input order: setup_inputs() insertion order, params dict recursively in
  // insertion order (fpi, cam, lid, cam_ts, lid_ts, tgt, then params leaves).
  auto IN = [&](int i) { return (const float*)d_in[i]; };
  const float* fpi    = IN(0);
  const float* cam    = IN(1);
  const float* lid    = IN(2);
  const float* cam_ts = IN(3);
  const float* lid_ts = IN(4);
  const float* tgt    = IN(5);
  // params (insertion order)
  const float *winit = IN(6),  *binit = IN(7);
  const float *wor = IN(8),  *bor = IN(9);
  const float *woz = IN(10), *boz = IN(11);
  const float *wog = IN(12), *bog = IN(13);
  const float *wjr = IN(14), *bjr = IN(15);
  const float *wjz = IN(16), *bjz = IN(17);
  const float *wjg = IN(18), *bjg = IN(19);
  const float *wout = IN(20), *bout = IN(21);
  const float *g0wu = IN(22), *g0bu = IN(23), *g0wr = IN(24), *g0br = IN(25), *g0wt = IN(26);
  const float *g1wu = IN(27), *g1bu = IN(28), *g1wr = IN(29), *g1br = IN(30), *g1wt = IN(31);
  const float *wb = IN(32), *bb = IN(33);
  const float *w0 = IN(34), *b0 = IN(35);
  const float *w1 = IN(36), *b1 = IN(37);
  const float *w2 = IN(38), *b2 = IN(39);
  const float *w3 = IN(40), *b3 = IN(41);
  const float *wgp = IN(42), *bgp = IN(43);
  const float *wproj = IN(44), *bproj = IN(45);
  const float *wh = IN(46), *bh = IN(47);
  const float *wlast = IN(48), *blast = IN(49);

  // ---- workspace layout (cursor allocator) ----
  char* W = (char*)d_ws;
  size_t cur = 0;
  auto alloc = [&](size_t bytes) {
    size_t o = cur;
    cur = (cur + bytes + 255) & ~(size_t)255;
    return o;
  };
  // schedule ints
  size_t o_has  = alloc((size_t)B_ * KST * 4);
  size_t o_src  = alloc((size_t)B_ * KST * 4);
  size_t o_psel = alloc((size_t)B_ * KST * 4);
  int* hasObs = (int*)(W + o_has);
  int* srcsel = (int*)(W + o_src);
  int* pselp  = (int*)(W + o_psel);

  // bf16 weights
  auto walloc = [&](int Co, int Ci, int T) { return alloc((size_t)Co * Ci * T * 2); };
  size_t ow_init = walloc(64, 64, 1);
  size_t ow_or = walloc(64, 64, 9), ow_oz = walloc(64, 64, 9), ow_og = walloc(64, 64, 9);
  size_t ow_jr = walloc(64, 128, 9), ow_jz = walloc(64, 128, 9), ow_jg = walloc(64, 128, 9);
  size_t ow_out = walloc(64, 64, 1);
  size_t ow_g0u = walloc(64, 128, 9), ow_g0r = walloc(64, 128, 9), ow_g0t = walloc(64, 128, 9);
  size_t ow_g1u = walloc(64, 128, 9), ow_g1r = walloc(64, 128, 9), ow_g1t = walloc(64, 128, 9);
  size_t ow_b  = walloc(64, 64, 9);
  size_t ow_0  = walloc(128, 64, 1);
  size_t ow_1  = walloc(128, 64, 9), ow_2 = walloc(128, 64, 9), ow_3 = walloc(128, 64, 9);
  size_t ow_pj = walloc(128, 512, 1);
  size_t ow_h  = walloc(128, 128, 9);
  size_t ow_l  = walloc(64, 128, 1);

  // float scratch
  auto falloc = [&](size_t elems) { return (float*)(W + alloc(elems * 4)); };
  float* h     = falloc((size_t)B_ * CI);
  float* bufR  = falloc((size_t)B_ * CI);
  float* bufZ  = falloc((size_t)B_ * CI);
  float* bufG  = falloc((size_t)B_ * CI);
  float* bufP  = falloc((size_t)B_ * CI);
  float* xh    = falloc((size_t)B_ * 2 * CI);
  float* preds = falloc((size_t)B_ * F_ * CI);
  float* sbuf  = falloc((size_t)B_ * CI);
  float* x1    = falloc((size_t)B_ * F_ * CI);
  float* yb    = falloc((size_t)B_ * F_ * CI);
  float* x2    = falloc((size_t)B_ * F_ * CI);
  float* bcat  = falloc((size_t)B_ * F_ * 4 * HID_ * IMG);   // 12 x 512 x 2304
  float* meanb = falloc((size_t)B_ * F_ * C_);
  float* gpb   = falloc((size_t)B_ * F_ * HID_);
  float* zb    = falloc((size_t)B_ * F_ * HID_);
  float* z1    = falloc((size_t)B_ * F_ * HID_ * IMG);
  float* z2    = falloc((size_t)B_ * F_ * HID_ * IMG);

  float* outx  = (float*)d_out;                       // (B,F,C,H,W)
  float* lossp = outx + (size_t)B_ * F_ * CI;         // scalar loss

  // ---- launch helpers ----
  auto prep = [&](const float* src, size_t off, int Co, int Ci, int ciOff,
                  int CinTot, int T) {
    int total = T * Ci * Co;
    k_prep_w<<<(total + 255) / 256, 256, 0, stream>>>(
        src, (__bf16*)(W + off), Co, Ci, ciOff, CinTot, T);
  };
  // templated dispatch on (Cin, taps); Cout is 64 or 128 (block N = 64)
  auto conv = [&](const float* X, long istr, float* Y, long ostr, size_t wOff,
                  const float* bias, int Cin, int Cout, int taps, int dil,
                  int act, int nImg, int biasPerImg = 0) {
    dim3 g(IMG / 64, Cout / 64, nImg), b(256);
    const __bf16* Wp = (const __bf16*)(W + wOff);
    if      (Cin ==  64 && taps == 1)
      k_conv_wmma< 64,1><<<g,b,0,stream>>>(X,istr,Y,ostr,Wp,bias,Cout,dil,act,biasPerImg);
    else if (Cin ==  64 && taps == 9)
      k_conv_wmma< 64,9><<<g,b,0,stream>>>(X,istr,Y,ostr,Wp,bias,Cout,dil,act,biasPerImg);
    else if (Cin == 128 && taps == 1)
      k_conv_wmma<128,1><<<g,b,0,stream>>>(X,istr,Y,ostr,Wp,bias,Cout,dil,act,biasPerImg);
    else if (Cin == 128 && taps == 9)
      k_conv_wmma<128,9><<<g,b,0,stream>>>(X,istr,Y,ostr,Wp,bias,Cout,dil,act,biasPerImg);
    else  /* Cin == 512, taps == 1 */
      k_conv_wmma<512,1><<<g,b,0,stream>>>(X,istr,Y,ostr,Wp,bias,Cout,dil,act,biasPerImg);
  };
  auto gr = [&](int n) { return dim3((n + 255) / 256); };
  const int NBC  = B_ * CI;
  const int NB2C = B_ * 2 * CI;

  // ---- weight conversion (once per call; tiny) ----
  prep(winit, ow_init, 64, 64, 0, 64, 1);
  prep(wor, ow_or, 64, 64, 0, 64, 9);
  prep(woz, ow_oz, 64, 64, 0, 64, 9);
  prep(wog, ow_og, 64, 64, 0, 64, 9);
  prep(wjr, ow_jr, 64, 128, 0, 128, 9);
  prep(wjz, ow_jz, 64, 128, 0, 128, 9);
  prep(wjg, ow_jg, 64, 128, 0, 128, 9);
  prep(wout, ow_out, 64, 64, 0, 64, 1);
  prep(g0wu, ow_g0u, 64, 128, 0, 128, 9);
  prep(g0wr, ow_g0r, 64, 128, 0, 128, 9);
  prep(g0wt, ow_g0t, 64, 128, 0, 128, 9);
  prep(g1wu, ow_g1u, 64, 128, 0, 128, 9);
  prep(g1wr, ow_g1r, 64, 128, 0, 128, 9);
  prep(g1wt, ow_g1t, 64, 128, 0, 128, 9);
  prep(wb, ow_b, 64, 64, 0, 64, 9);
  prep(w0, ow_0, 128, 64, 0, 64, 1);
  prep(w1, ow_1, 128, 64, 0, 64, 9);
  prep(w2, ow_2, 128, 64, 0, 64, 9);
  prep(w3, ow_3, 128, 64, 0, 64, 9);
  prep(wproj, ow_pj, 128, 512, 0, 640, 1);            // first 4 branches only
  prep(wh, ow_h, 128, 128, 0, 128, 9);
  prep(wlast, ow_l, 64, 128, 0, 128, 1);

  // ---- schedule + zero init ----
  k_schedule<<<1, 1, 0, stream>>>(cam_ts, lid_ts, tgt, hasObs, srcsel, pselp);
  hipMemsetAsync(preds, 0, (size_t)B_ * F_ * CI * 4, stream);
  hipMemsetAsync(lossp, 0, 4, stream);

  // ---- h = tanh(conv1x1(fpi, winit) + binit) ----
  conv(fpi, CI, h, CI, ow_init, binit, 64, 64, 1, 1, ACT_TANH, B_);

  // ---- K-step GRU-ODE scan ----
  for (int k = 0; k < KST; ++k) {
    conv(h, CI, bufR, CI, ow_or, bor, 64, 64, 9, 1, ACT_SIG, B_);
    conv(h, CI, bufZ, CI, ow_oz, boz, 64, 64, 9, 1, ACT_SIG, B_);
    k_mul<<<gr(NBC), 256, 0, stream>>>(bufP, bufR, h, NBC);        // r*h
    conv(bufP, CI, bufG, CI, ow_og, bog, 64, 64, 9, 1, ACT_TANH, B_);
    k_gruode<<<gr(NBC), 256, 0, stream>>>(h, bufZ, bufG, NBC);
    // p_pre + masked MSE loss
    conv(h, CI, bufP, CI, ow_out, bout, 64, 64, 1, 1, ACT_NONE, B_);
    k_loss<<<dim3((CI + 255) / 256, B_), 256, 0, stream>>>(
        bufP, cam, lid, hasObs, srcsel, k, lossp);
    // joint GRU update (masked per batch)
    k_build_xh<<<gr(NB2C), 256, 0, stream>>>(xh, cam, lid, h, srcsel, k);
    conv(xh, 2 * CI, bufZ, CI, ow_jz, bjz, 128, 64, 9, 1, ACT_SIG, B_);
    conv(xh, 2 * CI, bufR, CI, ow_jr, bjr, 128, 64, 9, 1, ACT_SIG, B_);
    k_xh_second<<<gr(NBC), 256, 0, stream>>>(xh, bufR, h);         // [xo, rj*h]
    conv(xh, 2 * CI, bufG, CI, ow_jg, bjg, 128, 64, 9, 1, ACT_TANH, B_);
    k_joint<<<gr(NBC), 256, 0, stream>>>(h, bufZ, bufG, hasObs, k);
    // prediction write (masked per batch)
    conv(h, CI, bufP, CI, ow_out, bout, 64, 64, 1, 1, ACT_NONE, B_);
    k_scatter_pred<<<gr(NBC), 256, 0, stream>>>(preds, bufP, pselp, k);
  }

  // ---- spatial GRU scan helper ----
  auto sgru = [&](const float* seq, float* out_seq, size_t owu, const float* bu,
                  size_t owr, const float* br, size_t owt) {
    k_copy_t0<<<gr(NBC), 256, 0, stream>>>(sbuf, preds);           // hidden = preds[:,0]
    for (int t = 0; t < F_; ++t) {
      k_build_xs<<<gr(NB2C), 256, 0, stream>>>(xh, seq, t, sbuf, nullptr, 0);
      conv(xh, 2 * CI, bufZ, CI, owu, bu, 128, 64, 9, 1, ACT_SIG, B_);   // u
      conv(xh, 2 * CI, bufR, CI, owr, br, 128, 64, 9, 1, ACT_SIG, B_);   // r
      k_build_xs<<<gr(NB2C), 256, 0, stream>>>(xh, seq, t, sbuf, bufR, 1);
      conv(xh, 2 * CI, bufG, CI, owt, nullptr, 128, 64, 9, 1, ACT_RELU, B_); // st
      k_sgru_update<<<gr(NBC), 256, 0, stream>>>(sbuf, out_seq, t, bufZ, bufG);
    }
  };

  sgru(preds, x1, ow_g0u, g0bu, ow_g0r, g0br, ow_g0t);
  conv(x1, CI, yb, CI, ow_b, bb, 64, 64, 9, 1, ACT_RELU, B_ * F_);
  sgru(yb, x2, ow_g1u, g1bu, ow_g1r, g1br, ow_g1t);

  // ---- ASPP-style head on 12 images ----
  const long OST = (long)4 * HID_ * IMG;                 // bcat image stride
  conv(x2, CI, bcat + 0 * HID_ * IMG, OST, ow_0, b0, 64, 128, 1, 1,  ACT_RELU, B_ * F_);
  conv(x2, CI, bcat + 1 * HID_ * IMG, OST, ow_1, b1, 64, 128, 9, 12, ACT_RELU, B_ * F_);
  conv(x2, CI, bcat + 2 * HID_ * IMG, OST, ow_2, b2, 64, 128, 9, 24, ACT_RELU, B_ * F_);
  conv(x2, CI, bcat + 3 * HID_ * IMG, OST, ow_3, b3, 64, 128, 9, 36, ACT_RELU, B_ * F_);
  // global-pool branch folded into per-image bias for wproj
  k_mean<<<dim3(B_ * F_), dim3(64), 0, stream>>>(x2, meanb);
  k_gp<<<gr(B_ * F_ * HID_), 256, 0, stream>>>(meanb, wgp, bgp, gpb);
  k_zbias<<<gr(B_ * F_ * HID_), 256, 0, stream>>>(gpb, wproj, bproj, zb);
  conv(bcat, OST, z1, (long)HID_ * IMG, ow_pj, zb, 512, 128, 1, 1, ACT_RELU,
       B_ * F_, /*biasPerImg=*/1);
  conv(z1, (long)HID_ * IMG, z2, (long)HID_ * IMG, ow_h, bh, 128, 128, 9, 1,
       ACT_RELU, B_ * F_);
  conv(z2, (long)HID_ * IMG, outx, CI, ow_l, blast, 128, 64, 1, 1, ACT_NONE,
       B_ * F_);
}
